// GotenNet_90606630077016
// MI455X (gfx1250) — matrix operand order repaired and checked by reading, not compiled, not run on hardware
//
#include <hip/hip_runtime.h>
#include <hip/hip_bf16.h>
#include <math.h>

// ---------------- constants ----------------
#define NN   16384
#define EE   131072
#define ZZ   128
#define HH   8

typedef float  v8f   __attribute__((ext_vector_type(8)));
typedef __bf16 v16bf __attribute__((ext_vector_type(16)));

// ---------------- helpers ----------------
__device__ __forceinline__ float siluf(float x) {
    return x * (1.0f / (1.0f + __expf(-x)));
}

// monotonic float <-> unsigned encoding for atomicMax on signed floats
__device__ __forceinline__ unsigned fenc(float f) {
    unsigned u = __float_as_uint(f);
    return (u & 0x80000000u) ? ~u : (u | 0x80000000u);
}
__device__ __forceinline__ float fdec(unsigned u) {
    return __uint_as_float((u & 0x80000000u) ? (u ^ 0x80000000u) : ~u);
}

// One 16x16 output tile, K=128 accumulated via 4x v_wmma_f32_16x16x32_bf16.
// A: LDS tile [16 rows][128 bf16] (row-major). B: packed weight fragments,
// layout [ntile][ktile][lane][16 bf16] (32 contiguous bytes per lane).
__device__ __forceinline__ v8f wmma_k128(const __bf16* A, const __bf16* B,
                                         int nt, v8f acc) {
    const int lane = threadIdx.x & 31;
    const int row  = lane & 15;
    const int hi   = lane >> 4;
#pragma unroll
    for (int kt = 0; kt < 4; ++kt) {
        union { v16bf v; uint4 u[2]; } a, b;
        const __bf16* ap = A + row * 128 + kt * 32 + hi * 16;
        a.u[0] = *reinterpret_cast<const uint4*>(ap);
        a.u[1] = *reinterpret_cast<const uint4*>(ap + 8);
        const __bf16* bp = B + (size_t)((nt * 4 + kt) * 32 + lane) * 16;
        b.u[0] = *reinterpret_cast<const uint4*>(bp);
        b.u[1] = *reinterpret_cast<const uint4*>(bp + 8);
        acc = __builtin_amdgcn_wmma_f32_16x16x32_bf16(
            false, a.v, false, b.v, (short)0, acc, false, false);
    }
    return acc;
}

// ---------------- weight packing ----------------
// W is [128 (K) x dout] row-major fp32.  Output: bf16 fragments,
// element t = (((nt*4 + kt)*32 + lane)*16 + j):
//   col n = nt*16 + lane%16 ; K = kt*32 + (lane>=16?16:0) + j
__global__ void kPack(const float* __restrict__ W, int dout,
                      __bf16* __restrict__ out) {
    int total = dout * 128;
    for (int t = blockIdx.x * blockDim.x + threadIdx.x; t < total;
         t += gridDim.x * blockDim.x) {
        int j    = t & 15;
        int lane = (t >> 4) & 31;
        int tile = t >> 9;
        int kt   = tile & 3;
        int nt   = tile >> 2;
        int n    = nt * 16 + (lane & 15);
        int kk   = kt * 32 + ((lane >> 4) ? 16 : 0) + j;
        out[t]   = (__bf16)W[(size_t)kk * dout + n];
    }
}

// ---------------- init ----------------
__global__ void kInit(const float* __restrict__ h, const float* __restrict__ mu,
                      float* __restrict__ out_h, float* __restrict__ out_mu,
                      unsigned* __restrict__ mmax, float* __restrict__ den) {
    long stride = (long)gridDim.x * blockDim.x;
    long t0     = (long)blockIdx.x * blockDim.x + threadIdx.x;
    for (long i = t0; i < (long)NN * 128; i += stride) out_h[i] = h[i];
    for (long i = t0; i < (long)NN * 384; i += stride) out_mu[i] = mu[i];
    for (long i = t0; i < (long)NN * 8; i += stride) { mmax[i] = 0u; den[i] = 0.0f; }
}

// ---------------- node-side fused MLPs ----------------
__global__ void __launch_bounds__(256)
kNode(const float* __restrict__ hs,
      const __bf16* pwq, const __bf16* pwk,
      const __bf16* pwgs1, const __bf16* pwgs2,
      const __bf16* pwgv1, const __bf16* pwgv2,
      const float* bq, const float* bk, const float* bgs1, const float* bgs2,
      const float* bgv1, const float* bgv2,
      float* __restrict__ q, float* __restrict__ k,
      float* __restrict__ x3, float* __restrict__ val3) {
    __shared__ __align__(16) __bf16 Ahs[16 * 128];
    __shared__ __align__(16) __bf16 At1[16 * 128];
    __shared__ __align__(16) __bf16 At2[16 * 128];

    const long row0 = (long)blockIdx.x * 16;
    for (int t = threadIdx.x; t < 16 * 128; t += 256)
        Ahs[t] = (__bf16)hs[row0 * 128 + t];
    __syncthreads();

    const int w    = threadIdx.x >> 5;
    const int lane = threadIdx.x & 31;
    const int n    = lane & 15;
    const int hi   = lane >> 4;

    // q, k : one 16-col tile per wave
    {
        v8f c = {};
        c = wmma_k128(Ahs, pwq, w, c);
        float bb = bq[w * 16 + n];
#pragma unroll
        for (int r = 0; r < 8; ++r)
            q[(row0 + hi * 8 + r) * 128 + w * 16 + n] = c[r] + bb;
    }
    {
        v8f c = {};
        c = wmma_k128(Ahs, pwk, w, c);
        float bb = bk[w * 16 + n];
#pragma unroll
        for (int r = 0; r < 8; ++r)
            k[(row0 + hi * 8 + r) * 128 + w * 16 + n] = c[r] + bb;
    }
    // hidden layers -> silu -> LDS (bf16, A layout = plain row-major [16][128])
    {
        v8f c = {};
        c = wmma_k128(Ahs, pwgs1, w, c);
        float bb = bgs1[w * 16 + n];
#pragma unroll
        for (int r = 0; r < 8; ++r)
            At1[(hi * 8 + r) * 128 + w * 16 + n] = (__bf16)siluf(c[r] + bb);
    }
    {
        v8f c = {};
        c = wmma_k128(Ahs, pwgv1, w, c);
        float bb = bgv1[w * 16 + n];
#pragma unroll
        for (int r = 0; r < 8; ++r)
            At2[(hi * 8 + r) * 128 + w * 16 + n] = (__bf16)siluf(c[r] + bb);
    }
    __syncthreads();

    // x3 / val3 : 24 col-tiles each, 3 per wave
#pragma unroll
    for (int i = 0; i < 3; ++i) {
        int nt = w * 3 + i;
        v8f c = {};
        c = wmma_k128(At1, pwgs2, nt, c);
        float bb = bgs2[nt * 16 + n];
#pragma unroll
        for (int r = 0; r < 8; ++r)
            x3[(row0 + hi * 8 + r) * 384 + nt * 16 + n] = c[r] + bb;
    }
#pragma unroll
    for (int i = 0; i < 3; ++i) {
        int nt = w * 3 + i;
        v8f c = {};
        c = wmma_k128(At2, pwgv2, nt, c);
        float bb = bgv2[nt * 16 + n];
#pragma unroll
        for (int r = 0; r < 8; ++r)
            val3[(row0 + hi * 8 + r) * 384 + nt * 16 + n] = c[r] + bb;
    }
}

// ---------------- edge pass 1: logits + segment max ----------------
__global__ void __launch_bounds__(256)
kLogits(const int* __restrict__ ei, const float* __restrict__ emb,
        const __bf16* pwphik, const float* bphik,
        const float* __restrict__ q, const float* __restrict__ k,
        float* __restrict__ logits, unsigned* __restrict__ mmax) {
    __shared__ __align__(16) __bf16 Aemb[16 * 128];
    __shared__ int sSrc[16], sDst[16];

    const long e0 = (long)blockIdx.x * 16;
    for (int t = threadIdx.x; t < 16 * 128; t += 256)
        Aemb[t] = (__bf16)emb[e0 * 128 + t];
    if (threadIdx.x < 16) {
        sSrc[threadIdx.x] = ei[e0 + threadIdx.x];
        sDst[threadIdx.x] = ei[(long)EE + e0 + threadIdx.x];
    }
    __syncthreads();

    const int w    = threadIdx.x >> 5;  // head
    const int lane = threadIdx.x & 31;
    const int n    = lane & 15;
    const int hi   = lane >> 4;

    v8f c = {};
    c = wmma_k128(Aemb, pwphik, w, c);  // r_attn pre-activation, head w
    float bb = bphik[w * 16 + n];
#pragma unroll
    for (int r = 0; r < 8; ++r) {
        int   m  = hi * 8 + r;           // edge within tile
        float ra = siluf(c[r] + bb);
        float qq = q[(long)sDst[m] * 128 + w * 16 + n];
        float kk = k[(long)sSrc[m] * 128 + w * 16 + n];
        float p  = ra * qq * kk;
#pragma unroll
        for (int off = 1; off < 16; off <<= 1) p += __shfl_xor(p, off);
        if (n == 0) {
            long e = e0 + m;
            logits[e * 8 + w] = p;
            atomicMax(&mmax[(long)sDst[m] * 8 + w], fenc(p));
        }
    }
}

// ---------------- edge pass 2: exp + denominators ----------------
__global__ void kDen(const int* __restrict__ ei, float* __restrict__ logits,
                     const unsigned* __restrict__ mmax, float* __restrict__ den) {
    long t = (long)blockIdx.x * blockDim.x + threadIdx.x;
    if (t >= (long)EE * 8) return;
    long e = t >> 3;
    int  h = (int)(t & 7);
    int  d = ei[(long)EE + e];
    float m  = fdec(mmax[(long)d * 8 + h]);
    float ex = __expf(logits[t] - m);
    logits[t] = ex;  // reuse buffer as ex
    atomicAdd(&den[(long)d * 8 + h], ex);
}

// ---------------- main edge kernel: r_ij + scatter ----------------
__global__ void __launch_bounds__(256)
kEdgeMain(const int* __restrict__ ei, const float* __restrict__ emb,
          const float* __restrict__ ew, const float* __restrict__ nedge,
          const float* __restrict__ evec, const float* __restrict__ mu,
          const __bf16* pwre, const float* bre,
          const float* __restrict__ exbuf, const float* __restrict__ den,
          const float* __restrict__ x3, const float* __restrict__ val3,
          float* __restrict__ out_h, float* __restrict__ out_mu) {
    __shared__ __align__(16) __bf16 Aemb[16 * 128];
    __shared__ float sAttn[16 * 8];
    __shared__ float sCut[16];
    __shared__ float sEv[16 * 3];
    __shared__ int   sSrc[16], sDst[16];

    const long e0 = (long)blockIdx.x * 16;
    for (int t = threadIdx.x; t < 16 * 128; t += 256)
        Aemb[t] = (__bf16)emb[e0 * 128 + t];
    if (threadIdx.x < 16) {
        int m = threadIdx.x;
        sSrc[m] = ei[e0 + m];
        sDst[m] = ei[(long)EE + e0 + m];
        float wgt = ew[e0 + m];
        sCut[m] = 0.5f * (cosf(3.14159265358979f * wgt / 5.0f) + 1.0f) *
                  (wgt < 5.0f ? 1.0f : 0.0f);
    }
    if (threadIdx.x < 48) sEv[threadIdx.x] = evec[e0 * 3 + threadIdx.x];
    if (threadIdx.x < 128) {
        int m = threadIdx.x >> 3, h = threadIdx.x & 7;
        long e  = e0 + m;
        int  d  = ei[(long)EE + e];
        float a = exbuf[e * 8 + h] / den[(long)d * 8 + h];
        sAttn[threadIdx.x] =
            a * sqrtf(nedge[e]) * (1.0f / sqrtf(128.0f));
    }
    __syncthreads();

    const int w    = threadIdx.x >> 5;
    const int lane = threadIdx.x & 31;
    const int n    = lane & 15;
    const int hi   = lane >> 4;

#pragma unroll
    for (int i = 0; i < 3; ++i) {
        int nt = w * 3 + i;                  // 0..23 -> col tile of 384
        v8f c = {};
        c = wmma_k128(Aemb, pwre, nt, c);    // r_ij tile
        int   cidx = nt * 16 + n;
        float bb   = bre[cidx];
        int   h    = cidx / 48;              // head for attention broadcast
#pragma unroll
        for (int r = 0; r < 8; ++r) {
            int  m   = hi * 8 + r;
            int  src = sSrc[m];
            int  dst = sDst[m];
            float rij = c[r] + bb;
            float a   = sAttn[m * 8 + h];
            float v3  = val3[(long)src * 384 + cidx];
            float xx3 = x3[(long)src * 384 + cidx];
            float xij = sCut[m] * a * v3 * xx3 * rij;
            if (cidx < 128) {
                atomicAdd(&out_h[(long)dst * 128 + cidx], xij);
            } else if (cidx < 256) {
                int cc = cidx - 128;         // o_d
#pragma unroll
                for (int l = 0; l < 3; ++l)
                    atomicAdd(&out_mu[(long)dst * 384 + l * 128 + cc],
                              xij * sEv[m * 3 + l]);
            } else {
                int cc = cidx - 256;         // o_t
#pragma unroll
                for (int l = 0; l < 3; ++l)
                    atomicAdd(&out_mu[(long)dst * 384 + l * 128 + cc],
                              xij * mu[(long)src * 384 + l * 128 + cc]);
            }
        }
    }
}

// ---------------- w1/w2 = mu_out @ {W_vq, W_vk0} ----------------
__global__ void __launch_bounds__(256)
kMuGemm(const float* __restrict__ mu_out, const __bf16* pwvq,
        const __bf16* pwvk0, float* __restrict__ w1, float* __restrict__ w2) {
    __shared__ __align__(16) __bf16 A[16 * 128];
    const long row0 = (long)blockIdx.x * 16;  // rows of [3N, 128]
    for (int t = threadIdx.x; t < 16 * 128; t += 256)
        A[t] = (__bf16)mu_out[row0 * 128 + t];
    __syncthreads();

    const int w    = threadIdx.x >> 5;
    const int lane = threadIdx.x & 31;
    const int n    = lane & 15;
    const int hi   = lane >> 4;
    {
        v8f c = {};
        c = wmma_k128(A, pwvq, w, c);
#pragma unroll
        for (int r = 0; r < 8; ++r)
            w1[(row0 + hi * 8 + r) * 128 + w * 16 + n] = c[r];
    }
    {
        v8f c = {};
        c = wmma_k128(A, pwvk0, w, c);
#pragma unroll
        for (int r = 0; r < 8; ++r)
            w2[(row0 + hi * 8 + r) * 128 + w * 16 + n] = c[r];
    }
}

// ---------------- final edge kernel: w_dot, f_up, df ----------------
__global__ void __launch_bounds__(256)
kFinal(const int* __restrict__ ei, const float* __restrict__ emb,
       const float* __restrict__ evec,
       const float* __restrict__ w1, const float* __restrict__ w2,
       const __bf16* pweu1, const __bf16* pweu2, const __bf16* pwlin,
       const float* beu1, const float* beu2, const float* blin,
       float* __restrict__ out_df) {
    __shared__ __align__(16) __bf16 Aemb[16 * 128];
    __shared__ __align__(16) __bf16 Ap[16 * 128];
    __shared__ __align__(16) __bf16 Ah[16 * 128];
    __shared__ float sEv[16 * 3];
    __shared__ int   sSrc[16], sDst[16];

    const long e0 = (long)blockIdx.x * 16;
    for (int t = threadIdx.x; t < 16 * 128; t += 256)
        Aemb[t] = (__bf16)emb[e0 * 128 + t];
    if (threadIdx.x < 16) {
        sSrc[threadIdx.x] = ei[e0 + threadIdx.x];
        sDst[threadIdx.x] = ei[(long)EE + e0 + threadIdx.x];
    }
    if (threadIdx.x < 48) sEv[threadIdx.x] = evec[e0 * 3 + threadIdx.x];
    __syncthreads();

    // p = (sum_l w1[dst][l]*ev[l]) * (sum_l w2[src][l]*ev[l]) -> bf16 LDS
    for (int t = threadIdx.x; t < 16 * 128; t += 256) {
        int m = t >> 7, c = t & 127;
        long db = (long)sDst[m] * 384, sb = (long)sSrc[m] * 384;
        float a0 = sEv[m * 3 + 0], a1 = sEv[m * 3 + 1], a2 = sEv[m * 3 + 2];
        float w1d = w1[db + c] * a0 + w1[db + 128 + c] * a1 + w1[db + 256 + c] * a2;
        float w2d = w2[sb + c] * a0 + w2[sb + 128 + c] * a1 + w2[sb + 256 + c] * a2;
        Ap[t] = (__bf16)(w1d * w2d);
    }

    const int w    = threadIdx.x >> 5;
    const int lane = threadIdx.x & 31;
    const int n    = lane & 15;
    const int hi   = lane >> 4;

    // hidden = silu(emb @ W_eu1 + b)
    {
        v8f c = {};
        c = wmma_k128(Aemb, pweu1, w, c);
        float bb = beu1[w * 16 + n];
#pragma unroll
        for (int r = 0; r < 8; ++r)
            Ah[(hi * 8 + r) * 128 + w * 16 + n] = (__bf16)siluf(c[r] + bb);
    }
    __syncthreads();

    // f_up and w_dot tiles, combine with emb into df
    v8f cf = {};
    cf = wmma_k128(Ah, pweu2, w, cf);
    v8f cw = {};
    cw = wmma_k128(Ap, pwlin, w, cw);
    float bf = beu2[w * 16 + n];
    float bl = blin[w * 16 + n];
#pragma unroll
    for (int r = 0; r < 8; ++r) {
        long e    = e0 + hi * 8 + r;
        int  cidx = w * 16 + n;
        float fup  = cf[r] + bf;
        float wdot = tanhf(cw[r] + bl);
        out_df[e * 128 + cidx] = emb[e * 128 + cidx] + fup * wdot;
    }
}

// ---------------- launch ----------------
extern "C" void kernel_launch(void* const* d_in, const int* in_sizes, int n_in,
                              void* d_out, int out_size, void* d_ws, size_t ws_size,
                              hipStream_t stream) {
    const int*   ei     = (const int*)d_in[0];
    const float* h      = (const float*)d_in[1];
    const float* mu     = (const float*)d_in[2];
    const float* evec   = (const float*)d_in[3];
    const float* emb    = (const float*)d_in[4];
    const float* ew     = (const float*)d_in[5];
    const float* nedge  = (const float*)d_in[6];
    const float* W_q    = (const float*)d_in[7];   const float* b_q   = (const float*)d_in[8];
    const float* W_k    = (const float*)d_in[9];   const float* b_k   = (const float*)d_in[10];
    const float* W_gs1  = (const float*)d_in[11];  const float* b_gs1 = (const float*)d_in[12];
    const float* W_gs2  = (const float*)d_in[13];  const float* b_gs2 = (const float*)d_in[14];
    const float* W_gv1  = (const float*)d_in[15];  const float* b_gv1 = (const float*)d_in[16];
    const float* W_gv2  = (const float*)d_in[17];  const float* b_gv2 = (const float*)d_in[18];
    const float* W_phik = (const float*)d_in[19];  const float* b_phik= (const float*)d_in[20];
    const float* W_re   = (const float*)d_in[21];  const float* b_re  = (const float*)d_in[22];
    const float* W_vq   = (const float*)d_in[23];
    const float* W_vk0  = (const float*)d_in[24];
    const float* W_lin  = (const float*)d_in[25];  const float* b_lin = (const float*)d_in[26];
    const float* W_eu1  = (const float*)d_in[27];  const float* b_eu1 = (const float*)d_in[28];
    const float* W_eu2  = (const float*)d_in[29];  const float* b_eu2 = (const float*)d_in[30];

    float* out_h  = (float*)d_out;
    float* out_mu = out_h + (size_t)NN * 128;
    float* out_df = out_mu + (size_t)NN * 384;

    // workspace carve-up (256B aligned)
    char*  ws  = (char*)d_ws;
    size_t off = 0;
    auto alloc = [&](size_t bytes) {
        void* p = ws + off;
        off = (off + bytes + 255) & ~(size_t)255;
        return p;
    };
    __bf16* pwq    = (__bf16*)alloc((size_t)128 * 128 * 2);
    __bf16* pwk    = (__bf16*)alloc((size_t)128 * 128 * 2);
    __bf16* pwgs1  = (__bf16*)alloc((size_t)128 * 128 * 2);
    __bf16* pwgs2  = (__bf16*)alloc((size_t)128 * 384 * 2);
    __bf16* pwgv1  = (__bf16*)alloc((size_t)128 * 128 * 2);
    __bf16* pwgv2  = (__bf16*)alloc((size_t)128 * 384 * 2);
    __bf16* pwphik = (__bf16*)alloc((size_t)128 * 128 * 2);
    __bf16* pwre   = (__bf16*)alloc((size_t)128 * 384 * 2);
    __bf16* pwvq   = (__bf16*)alloc((size_t)128 * 128 * 2);
    __bf16* pwvk0  = (__bf16*)alloc((size_t)128 * 128 * 2);
    __bf16* pwlin  = (__bf16*)alloc((size_t)128 * 128 * 2);
    __bf16* pweu1  = (__bf16*)alloc((size_t)128 * 128 * 2);
    __bf16* pweu2  = (__bf16*)alloc((size_t)128 * 128 * 2);
    float*    q      = (float*)alloc((size_t)NN * 128 * 4);
    float*    k      = (float*)alloc((size_t)NN * 128 * 4);
    float*    x3     = (float*)alloc((size_t)NN * 384 * 4);
    float*    val3   = (float*)alloc((size_t)NN * 384 * 4);
    float*    logits = (float*)alloc((size_t)EE * 8 * 4);
    unsigned* mmax   = (unsigned*)alloc((size_t)NN * 8 * 4);
    float*    den    = (float*)alloc((size_t)NN * 8 * 4);
    float*    w1     = (float*)alloc((size_t)NN * 384 * 4);
    float*    w2     = (float*)alloc((size_t)NN * 384 * 4);
    (void)ws_size; (void)n_in; (void)in_sizes; (void)out_size;

    // 1) pack weights to bf16 WMMA fragments
    auto pack = [&](const float* W, int dout, __bf16* dst) {
        int total = dout * 128;
        kPack<<<(total + 255) / 256, 256, 0, stream>>>(W, dout, dst);
    };
    pack(W_q, 128, pwq);     pack(W_k, 128, pwk);
    pack(W_gs1, 128, pwgs1); pack(W_gs2, 384, pwgs2);
    pack(W_gv1, 128, pwgv1); pack(W_gv2, 384, pwgv2);
    pack(W_phik, 128, pwphik); pack(W_re, 384, pwre);
    pack(W_vq, 128, pwvq);   pack(W_vk0, 128, pwvk0);
    pack(W_lin, 128, pwlin); pack(W_eu1, 128, pweu1); pack(W_eu2, 128, pweu2);

    // 2) init outputs (residual bases) and softmax accumulators
    kInit<<<2048, 256, 0, stream>>>(h, mu, out_h, out_mu, mmax, den);

    // 3) node-side fused MLPs (WMMA)
    kNode<<<NN / 16, 256, 0, stream>>>(h, pwq, pwk, pwgs1, pwgs2, pwgv1, pwgv2,
                                       b_q, b_k, b_gs1, b_gs2, b_gv1, b_gv2,
                                       q, k, x3, val3);

    // 4) logits + segment max
    kLogits<<<EE / 16, 256, 0, stream>>>(ei, emb, pwphik, b_phik, q, k,
                                         logits, mmax);

    // 5) exp + denominators
    kDen<<<(EE * 8) / 256, 256, 0, stream>>>(ei, logits, mmax, den);

    // 6) main edge scatter (r_ij via WMMA; atomics into out_h/out_mu)
    kEdgeMain<<<EE / 16, 256, 0, stream>>>(ei, emb, ew, nedge, evec, mu,
                                           pwre, b_re, logits, den, x3, val3,
                                           out_h, out_mu);

    // 7) w1/w2 = mu_out @ {W_vq, W_vk0}
    kMuGemm<<<(3 * NN) / 16, 256, 0, stream>>>(out_mu, pwvq, pwvk0, w1, w2);

    // 8) final edge kernel: f_up, w_dot, df
    kFinal<<<EE / 16, 256, 0, stream>>>(ei, emb, evec, w1, w2,
                                        pweu1, pweu2, pwlin,
                                        b_eu1, b_eu2, b_lin, out_df);
}
